// LineVectorizer_51256139711100
// MI455X (gfx1250) — matrix operand order repaired and checked by reading, not compiled, not run on hardware
//
#include <hip/hip_runtime.h>
#include <hip/hip_bf16.h>
#include <math.h>

typedef __attribute__((ext_vector_type(16))) __bf16 v16bf;
typedef __attribute__((ext_vector_type(8)))  float  v8f;

#define HH    128
#define WW    128
#define HWSZ  (HH*WW)
#define NB    8
#define NL    512
#define CIN   256
#define DLOI  128
#define NP0   64
#define NP1   32
#define FDIM  8
#define DFC   1024
#define KRAW  4104        // DLOI*NP1 + FDIM
#define KPAD  4128        // padded to multiple of 32

// workspace offsets (bytes)
#define OFF_X     0ull            // f32  [8][128][16384]            = 67,108,864 B
#define OFF_H1B   0ull            // bf16 [4096][1024] (reuses X)    =  8,388,608 B
#define OFF_H2    16777216ull     // f32  [4096][1024] (reuses X)    = 16,777,216 B
#define OFF_XCAT  67108864ull     // bf16 [4096][4128]               = 33,816,576 B
#define OFF_W1B   100925440ull    // bf16 [1024][4128]               =  8,454,144 B
#define OFF_W2B   109379584ull    // bf16 [1024][1024]               =  2,097,152 B

// ---------------------------------------------------------------------------
// WMMA tile loaders.
// A 16x32 bf16 layout: lane l: m=l&15, half=l>>4.
//   elements 0..7  = K[half*8 + 0..7],  elements 8..15 = K[16 + half*8 + 0..7]
// B 32x16 bf16 layout: lane l: n=l&15, half=l>>4.
//   elements 0..7  = K[half*8 + 0..7],  elements 8..15 = K[16 + half*8 + 0..7]
// ---------------------------------------------------------------------------
__device__ __forceinline__ v16bf ldA16(const __bf16* p /* row + k0 + half*8 */) {
    union U { uint4 q; __bf16 h[8]; } a, b;
    a.q = *reinterpret_cast<const uint4*>(p);
    b.q = *reinterpret_cast<const uint4*>(p + 16);
    v16bf r;
#pragma unroll
    for (int i = 0; i < 8; ++i) { r[i] = a.h[i]; r[i + 8] = b.h[i]; }
    return r;
}

__device__ __forceinline__ v16bf ldB16(const __bf16* p /* row + k0 + half*8 */) {
    union U { uint4 q; __bf16 h[8]; } a, b;
    a.q = *reinterpret_cast<const uint4*>(p);
    b.q = *reinterpret_cast<const uint4*>(p + 16);
    v16bf r;
#pragma unroll
    for (int i = 0; i < 8; ++i) { r[i] = a.h[i]; r[i + 8] = b.h[i]; }
    return r;
}

// ---------------------------------------------------------------------------
// Prep: w1 -> bf16 padded [1024][4128], w2 -> bf16 [1024][1024]
// ---------------------------------------------------------------------------
__global__ __launch_bounds__(256) void prep_weights(const float* __restrict__ w1,
                                                    const float* __restrict__ w2,
                                                    __bf16* __restrict__ w1b,
                                                    __bf16* __restrict__ w2b) {
    int i = blockIdx.x * 256 + threadIdx.x;
    const int n1 = DFC * KPAD;
    const int n2 = DFC * DFC;
    if (i < n1) {
        int r = i / KPAD, k = i - r * KPAD;
        w1b[i] = (__bf16)((k < KRAW) ? w1[(size_t)r * KRAW + k] : 0.0f);
    } else if (i < n1 + n2) {
        int j = i - n1;
        w2b[j] = (__bf16)w2[j];
    }
}

// ---------------------------------------------------------------------------
// fc1 GEMM fused with bias + attention: x[b][m][p] = (W f + b)[m,p] * (1+lmap[b][p])
// M=128, N=16384 per batch, K=256.
// One wave computes ALL 8 M-tiles for a 16-pixel tile, so the expensive
// channel-strided feature B-tile is loaded once and reused by 8 WMMAs.
// ---------------------------------------------------------------------------
__global__ __launch_bounds__(256) void fc1_gemm(const float* __restrict__ W,
                                                const float* __restrict__ bias,
                                                const float* __restrict__ feature,
                                                const float* __restrict__ lmap,
                                                float* __restrict__ x) {
    int wid  = (blockIdx.x * blockDim.x + threadIdx.x) >> 5;  // 8192 waves
    int lane = threadIdx.x & 31;
    int pt = wid & 1023;      // pixel tile (16 px)
    int b  = wid >> 10;       // batch

    int r = lane & 15, half = lane >> 4;
    const float* fbase = feature + (size_t)b * CIN * HWSZ + (size_t)(pt * 16 + r);

    v8f acc[8];
#pragma unroll
    for (int mt = 0; mt < 8; ++mt)
        acc[mt] = (v8f){0.f, 0.f, 0.f, 0.f, 0.f, 0.f, 0.f, 0.f};

    for (int k0 = 0; k0 < CIN; k0 += 32) {
        // B tile: strided channel loads (shared by all 8 M-tiles)
        int kB = k0 + half * 8;
        v16bf bt;
#pragma unroll
        for (int i = 0; i < 16; ++i)
            bt[i] = (__bf16)fbase[(size_t)(kB + (i < 8 ? i : i + 8)) * HWSZ];

#pragma unroll
        for (int mt = 0; mt < 8; ++mt) {
            const float* arow = W + (size_t)(mt * 16 + r) * CIN + k0 + half * 8;
            union F4 { float4 v; float f[4]; } a0, a1, a2, a3;
            a0.v = *reinterpret_cast<const float4*>(arow);
            a1.v = *reinterpret_cast<const float4*>(arow + 4);
            a2.v = *reinterpret_cast<const float4*>(arow + 16);
            a3.v = *reinterpret_cast<const float4*>(arow + 20);
            v16bf at;
#pragma unroll
            for (int i = 0; i < 4; ++i) {
                at[i]      = (__bf16)a0.f[i];
                at[4 + i]  = (__bf16)a1.f[i];
                at[8 + i]  = (__bf16)a2.f[i];
                at[12 + i] = (__bf16)a3.f[i];
            }
            acc[mt] = __builtin_amdgcn_wmma_f32_16x16x32_bf16(false, at, false, bt,
                                                              (short)0, acc[mt],
                                                              false, false);
        }
    }

    int p = pt * 16 + (lane & 15);
    float att = 1.0f + lmap[(size_t)b * HWSZ + p];
    float* xb = x + (size_t)b * DLOI * HWSZ;
#pragma unroll
    for (int mt = 0; mt < 8; ++mt) {
        int m0 = mt * 16 + half * 8;
#pragma unroll
        for (int j = 0; j < 8; ++j)
            xb[(size_t)(m0 + j) * HWSZ + p] = (acc[mt][j] + bias[m0 + j]) * att;
    }
}

// ---------------------------------------------------------------------------
// Per-line kernel: points, bilinear xa, conv1d stack, fcline mask, xp sample,
// mask + maxpool(2) -> xcat row (bf16, padded to 4128)
// ---------------------------------------------------------------------------
__global__ __launch_bounds__(256) void line_kernel(
    const float* __restrict__ x, const float* __restrict__ lmap,
    const float* __restrict__ lines, const float* __restrict__ feat,
    const float* __restrict__ c1w, const float* __restrict__ c1b,
    const float* __restrict__ s1,  const float* __restrict__ bb1,
    const float* __restrict__ c2w, const float* __restrict__ c2b,
    const float* __restrict__ s2,  const float* __restrict__ bb2,
    const float* __restrict__ c3w, const float* __restrict__ c3b,
    const float* __restrict__ s3,  const float* __restrict__ bb3,
    const float* __restrict__ flw, const float* __restrict__ flb,
    __bf16* __restrict__ xcat) {
    int row = blockIdx.x;          // 0..4095
    int b = row / NL;
    int tid = threadIdx.x;

    __shared__ int   s_idx[NP0 * 4];
    __shared__ float s_w[NP0 * 4];
    __shared__ float s_xa[NP0];
    __shared__ float s_h[2][64 * 65];
    __shared__ float s_red[256];
    __shared__ float s_keep[NP0];
    __shared__ float s_fb[2];

    // ---- points + attention sample ----
    if (tid < NP0) {
        float lam = (float)tid * (1.0f / 63.0f);
        const float* lp = lines + (size_t)row * 4;
        float px = lp[0] * lam + lp[2] * (1.0f - lam) - 0.5f;
        float py = lp[1] * lam + lp[3] * (1.0f - lam) - 0.5f;
        float px0 = fminf(fmaxf(floorf(px), 0.0f), 127.0f);
        float py0 = fminf(fmaxf(floorf(py), 0.0f), 127.0f);
        float px1 = fminf(px0 + 1.0f, 127.0f);
        float py1 = fminf(py0 + 1.0f, 127.0f);
        int i00 = (int)(px0 * 128.0f + py0);
        int i10 = (int)(px1 * 128.0f + py0);
        int i01 = (int)(px0 * 128.0f + py1);
        int i11 = (int)(px1 * 128.0f + py1);
        float w00 = (px1 - px) * (py1 - py);
        float w10 = (px - px0) * (py1 - py);
        float w01 = (px1 - px) * (py - py0);
        float w11 = (px - px0) * (py - py0);
        s_idx[tid * 4 + 0] = i00; s_idx[tid * 4 + 1] = i10;
        s_idx[tid * 4 + 2] = i01; s_idx[tid * 4 + 3] = i11;
        s_w[tid * 4 + 0] = w00;   s_w[tid * 4 + 1] = w10;
        s_w[tid * 4 + 2] = w01;   s_w[tid * 4 + 3] = w11;
        const float* lm = lmap + (size_t)b * HWSZ;
        s_xa[tid] = lm[i00] * w00 + lm[i10] * w10 + lm[i01] * w01 + lm[i11] * w11;
    }
    __syncthreads();

    // ---- conv1 (1 -> 64 ch) + bn + relu ----
    for (int j = tid; j < 64 * 64; j += 256) {
        int o = j >> 6, t = j & 63;
        float xl = (t > 0)  ? s_xa[t - 1] : 0.0f;
        float xc = s_xa[t];
        float xr = (t < 63) ? s_xa[t + 1] : 0.0f;
        float v = c1w[o * 3 + 0] * xl + c1w[o * 3 + 1] * xc + c1w[o * 3 + 2] * xr;
        v = (v + c1b[o]) * s1[o] + bb1[o];
        s_h[0][o * 65 + t] = fmaxf(v, 0.0f);
    }
    __syncthreads();

    // ---- conv2 (64 -> 64) ----
    for (int j = tid; j < 64 * 64; j += 256) {
        int o = j >> 6, t = j & 63;
        float acc = c2b[o];
        const float* wrow = c2w + o * 192;
        for (int c = 0; c < 64; ++c) {
            float hl = (t > 0)  ? s_h[0][c * 65 + t - 1] : 0.0f;
            float hc = s_h[0][c * 65 + t];
            float hr = (t < 63) ? s_h[0][c * 65 + t + 1] : 0.0f;
            acc += wrow[c * 3] * hl + wrow[c * 3 + 1] * hc + wrow[c * 3 + 2] * hr;
        }
        acc = acc * s2[o] + bb2[o];
        s_h[1][o * 65 + t] = fmaxf(acc, 0.0f);
    }
    __syncthreads();

    // ---- conv3 (64 -> 64) ----
    for (int j = tid; j < 64 * 64; j += 256) {
        int o = j >> 6, t = j & 63;
        float acc = c3b[o];
        const float* wrow = c3w + o * 192;
        for (int c = 0; c < 64; ++c) {
            float hl = (t > 0)  ? s_h[1][c * 65 + t - 1] : 0.0f;
            float hc = s_h[1][c * 65 + t];
            float hr = (t < 63) ? s_h[1][c * 65 + t + 1] : 0.0f;
            acc += wrow[c * 3] * hl + wrow[c * 3 + 1] * hc + wrow[c * 3 + 2] * hr;
        }
        acc = acc * s3[o] + bb3[o];
        s_h[0][o * 65 + t] = fmaxf(acc, 0.0f);
    }
    __syncthreads();

    // ---- fcline: two dots over 4096 + sigmoid + ceil ----
    float d0 = 0.0f, d1 = 0.0f;
    for (int j = tid; j < 4096; j += 256) {
        int o = j >> 6, t = j & 63;
        float hv = s_h[0][o * 65 + t];
        d0 += hv * flw[j];
        d1 += hv * flw[4096 + j];
    }
    s_red[tid] = d0; __syncthreads();
    for (int s = 128; s > 0; s >>= 1) {
        if (tid < s) s_red[tid] += s_red[tid + s];
        __syncthreads();
    }
    if (tid == 0) {
        float z = s_red[0] + flb[0];
        s_fb[0] = ceilf((1.0f / (1.0f + expf(-z))) * 64.0f);
    }
    __syncthreads();
    s_red[tid] = d1; __syncthreads();
    for (int s = 128; s > 0; s >>= 1) {
        if (tid < s) s_red[tid] += s_red[tid + s];
        __syncthreads();
    }
    if (tid == 0) {
        float z = s_red[0] + flb[1];
        s_fb[1] = ceilf((1.0f / (1.0f + expf(-z))) * 64.0f);
    }
    __syncthreads();
    if (tid < NP0) {
        float r = (float)tid;
        bool zero = (r >= s_fb[0]) && (r < s_fb[1]);
        s_keep[tid] = zero ? 0.0f : 1.0f;
    }
    __syncthreads();

    // ---- xp sample, mask, maxpool(2), store bf16 xcat row ----
    const float* xb = x + (size_t)b * DLOI * HWSZ;
    __bf16* xrow = xcat + (size_t)row * KPAD;
    for (int j = tid; j < DLOI * NP1; j += 256) {
        int c = j >> 5, s = j & 31;
        const float* xc_ = xb + (size_t)c * HWSZ;
        int t0 = 2 * s, t1 = 2 * s + 1;
        float v0 = (xc_[s_idx[t0 * 4 + 0]] * s_w[t0 * 4 + 0] +
                    xc_[s_idx[t0 * 4 + 1]] * s_w[t0 * 4 + 1] +
                    xc_[s_idx[t0 * 4 + 2]] * s_w[t0 * 4 + 2] +
                    xc_[s_idx[t0 * 4 + 3]] * s_w[t0 * 4 + 3]) * s_keep[t0];
        float v1 = (xc_[s_idx[t1 * 4 + 0]] * s_w[t1 * 4 + 0] +
                    xc_[s_idx[t1 * 4 + 1]] * s_w[t1 * 4 + 1] +
                    xc_[s_idx[t1 * 4 + 2]] * s_w[t1 * 4 + 2] +
                    xc_[s_idx[t1 * 4 + 3]] * s_w[t1 * 4 + 3]) * s_keep[t1];
        xrow[c * 32 + s] = (__bf16)fmaxf(v0, v1);
    }
    if (tid < 32) {
        float fv = (tid < FDIM) ? feat[(size_t)row * FDIM + tid] : 0.0f;
        xrow[DLOI * NP1 + tid] = (__bf16)fv;   // feat + zero pad to 4128
    }
}

// ---------------------------------------------------------------------------
// Generic NT GEMM, bf16 x bf16 -> f32, fused bias (+relu), out f32 or bf16.
// D[m][n] = sum_k A[m][k]*B[n][k]. One wave computes a 16M x 64N block
// (4 accumulators): the A tile is loaded once per k-step and reused by 4 WMMAs.
// Grid covers tiles exactly (EXEC all-ones at every WMMA).
// ---------------------------------------------------------------------------
__global__ __launch_bounds__(256) void gemm_nt_bf16(
    const __bf16* __restrict__ A, const __bf16* __restrict__ B,
    const float* __restrict__ bias, int mtiles, int K, int lda, int ldb,
    int ldo, int relu, float* __restrict__ outF, __bf16* __restrict__ outH) {
    int wid  = (blockIdx.x * blockDim.x + threadIdx.x) >> 5;
    int lane = threadIdx.x & 31;
    int mt = wid % mtiles;        // 16-row M tile
    int ng = wid / mtiles;        // 64-col N group
    int n0 = ng * 64;
    int r = lane & 15, half = lane >> 4;

    const __bf16* arow  = A + (size_t)(mt * 16 + r) * lda;
    const __bf16* brow0 = B + (size_t)(n0 + 0 * 16 + r) * ldb;
    const __bf16* brow1 = B + (size_t)(n0 + 1 * 16 + r) * ldb;
    const __bf16* brow2 = B + (size_t)(n0 + 2 * 16 + r) * ldb;
    const __bf16* brow3 = B + (size_t)(n0 + 3 * 16 + r) * ldb;

    v8f acc0 = {0.f, 0.f, 0.f, 0.f, 0.f, 0.f, 0.f, 0.f};
    v8f acc1 = acc0, acc2 = acc0, acc3 = acc0;

    for (int k0 = 0; k0 < K; k0 += 32) {
        v16bf at = ldA16(arow + k0 + half * 8);
        int kB = k0 + half * 8;
        v16bf bt0 = ldB16(brow0 + kB);
        v16bf bt1 = ldB16(brow1 + kB);
        v16bf bt2 = ldB16(brow2 + kB);
        v16bf bt3 = ldB16(brow3 + kB);
        acc0 = __builtin_amdgcn_wmma_f32_16x16x32_bf16(false, at, false, bt0,
                                                       (short)0, acc0, false, false);
        acc1 = __builtin_amdgcn_wmma_f32_16x16x32_bf16(false, at, false, bt1,
                                                       (short)0, acc1, false, false);
        acc2 = __builtin_amdgcn_wmma_f32_16x16x32_bf16(false, at, false, bt2,
                                                       (short)0, acc2, false, false);
        acc3 = __builtin_amdgcn_wmma_f32_16x16x32_bf16(false, at, false, bt3,
                                                       (short)0, acc3, false, false);
    }

    int m0 = mt * 16 + half * 8;
    float bv[8];
#pragma unroll
    for (int j = 0; j < 8; ++j) bv[j] = bias[m0 + j];

    v8f accs[4] = {acc0, acc1, acc2, acc3};
#pragma unroll
    for (int s = 0; s < 4; ++s) {
        int n = n0 + s * 16 + (lane & 15);
        float v[8];
#pragma unroll
        for (int j = 0; j < 8; ++j) {
            float t = accs[s][j] + bv[j];
            v[j] = relu ? fmaxf(t, 0.0f) : t;
        }
        if (outH) {
            union { uint4 q; __bf16 h[8]; } o;
#pragma unroll
            for (int j = 0; j < 8; ++j) o.h[j] = (__bf16)v[j];
            *reinterpret_cast<uint4*>(outH + (size_t)n * ldo + m0) = o.q;
        } else {
            float* p = outF + (size_t)n * ldo + m0;
            *reinterpret_cast<float4*>(p)     = make_float4(v[0], v[1], v[2], v[3]);
            *reinterpret_cast<float4*>(p + 4) = make_float4(v[4], v[5], v[6], v[7]);
        }
    }
}

// ---------------------------------------------------------------------------
// Final dot: out[n] = h2[n] . w3 + b3  (one wave per row)
// ---------------------------------------------------------------------------
__global__ __launch_bounds__(256) void final_dot(const float* __restrict__ h2,
                                                 const float* __restrict__ w3,
                                                 const float* __restrict__ b3,
                                                 float* __restrict__ out) {
    int wid  = (blockIdx.x * blockDim.x + threadIdx.x) >> 5;  // row 0..4095
    int lane = threadIdx.x & 31;
    const float* rowp = h2 + (size_t)wid * DFC;
    float s = 0.0f;
    for (int k = lane; k < DFC; k += 32) s += rowp[k] * w3[k];
#pragma unroll
    for (int off = 16; off > 0; off >>= 1) s += __shfl_down(s, off, 32);
    if (lane == 0) out[wid] = s + b3[0];
}

// ---------------------------------------------------------------------------
extern "C" void kernel_launch(void* const* d_in, const int* in_sizes, int n_in,
                              void* d_out, int out_size, void* d_ws, size_t ws_size,
                              hipStream_t stream) {
    (void)in_sizes; (void)n_in; (void)out_size; (void)ws_size;
    const float* feature = (const float*)d_in[0];
    const float* lmap    = (const float*)d_in[1];
    const float* lines   = (const float*)d_in[2];
    const float* feat    = (const float*)d_in[3];
    const float* fc1_w   = (const float*)d_in[4];
    const float* fc1_b   = (const float*)d_in[5];
    const float* c1w     = (const float*)d_in[6];
    const float* c1b     = (const float*)d_in[7];
    const float* bn1s    = (const float*)d_in[8];
    const float* bn1b    = (const float*)d_in[9];
    const float* c2w     = (const float*)d_in[10];
    const float* c2b     = (const float*)d_in[11];
    const float* bn2s    = (const float*)d_in[12];
    const float* bn2b    = (const float*)d_in[13];
    const float* c3w     = (const float*)d_in[14];
    const float* c3b     = (const float*)d_in[15];
    const float* bn3s    = (const float*)d_in[16];
    const float* bn3b    = (const float*)d_in[17];
    const float* flw     = (const float*)d_in[18];
    const float* flb     = (const float*)d_in[19];
    const float* w1      = (const float*)d_in[20];
    const float* b1      = (const float*)d_in[21];
    const float* w2      = (const float*)d_in[22];
    const float* b2      = (const float*)d_in[23];
    const float* w3      = (const float*)d_in[24];
    const float* b3      = (const float*)d_in[25];

    char* ws = (char*)d_ws;
    float*  x    = (float*)(ws + OFF_X);
    __bf16* xcat = (__bf16*)(ws + OFF_XCAT);
    __bf16* w1b  = (__bf16*)(ws + OFF_W1B);
    __bf16* w2b  = (__bf16*)(ws + OFF_W2B);
    __bf16* h1b  = (__bf16*)(ws + OFF_H1B);
    float*  h2   = (float*)(ws + OFF_H2);
    float*  out  = (float*)d_out;

    // 1. weight conversion / padding
    {
        int total = DFC * KPAD + DFC * DFC;
        int blocks = (total + 255) / 256;
        prep_weights<<<blocks, 256, 0, stream>>>(w1, w2, w1b, w2b);
    }
    // 2. fc1 GEMM (+bias, +attention): 1024 pixel-tiles * 8 batches = 8192 waves
    fc1_gemm<<<1024, 256, 0, stream>>>(fc1_w, fc1_b, feature, lmap, x);
    // 3. per-line sampling + conv stack + mask + pool -> xcat (bf16)
    line_kernel<<<NB * NL, 256, 0, stream>>>(x, lmap, lines, feat,
                                             c1w, c1b, bn1s, bn1b,
                                             c2w, c2b, bn2s, bn2b,
                                             c3w, c3b, bn3s, bn3b,
                                             flw, flb, xcat);
    // 4. MLP layer 1: M=1024 (64 mtiles), N=4096 (64 groups of 64) -> h1 (bf16)
    gemm_nt_bf16<<<(64 * 64) / 8, 256, 0, stream>>>(w1b, xcat, b1,
                                                    64, KPAD, KPAD, KPAD,
                                                    DFC, 1, nullptr, h1b);
    // 5. MLP layer 2: M=1024, N=4096, K=1024 -> h2 (f32)
    gemm_nt_bf16<<<(64 * 64) / 8, 256, 0, stream>>>(w2b, h1b, b2,
                                                    64, DFC, DFC, DFC,
                                                    DFC, 1, h2, nullptr);
    // 6. final dot: 4096 rows, wave per row
    final_dot<<<4096 / 8, 256, 0, stream>>>(h2, w3, b3, out);
}
